// CavAttention_62062277427840
// MI455X (gfx1250) — compile-verified, matrix operand order checked by reading, not run on hardware
//
#include <hip/hip_runtime.h>
#include <hip/hip_bf16.h>

typedef __attribute__((ext_vector_type(16))) __bf16 v16bf;
typedef __attribute__((ext_vector_type(8)))  __bf16 v8bf;
typedef __attribute__((ext_vector_type(8)))  float  v8f;

#define BB     2
#define LL     5
#define HH     100
#define WWW    176
#define CC     256
#define HWHW   (HH * WWW)          /* 17600 */
#define SITES  (BB * HWHW)         /* 35200 */
#define SITES_PER_BLK 16
#define ROWS   (SITES_PER_BLK * LL) /* 80 = 5 WMMA M-tiles */
#define LDA    264                  /* padded ld (halves) for 80x256 bf16 tile */
#define LDQF   772                  /* padded ld (floats) for 80x768 f32 qkv   */
#define LDO    260                  /* padded ld (floats) for 80x256 f32 out   */
#define NQKV   768
#define SMEM_A_BYTES  (ROWS * LDA * 2)          /* 42,240  */
#define SMEM_Q_BYTES  (ROWS * LDQF * 4)         /* 247,040 */
#define SMEM_BYTES    (SMEM_A_BYTES + SMEM_Q_BYTES) /* 289,280 < 320K */
#define SCALEF 0.17677669529663689f /* 32^-0.5 */

// Load a 16-half WMMA fragment: halves [0..7] and [16..23] relative to p
// (ISA 16-bit A/B layout: lanes<16 -> K 0..7 & 16..23, lanes>=16 -> +8).
__device__ __forceinline__ v16bf load_frag16(const __bf16* p) {
    v8bf lo = *(const v8bf*)p;
    v8bf hi = *(const v8bf*)(p + 16);
    v16bf r;
#pragma unroll
    for (int i = 0; i < 8; ++i) { r[i] = lo[i]; r[i + 8] = hi[i]; }
    return r;
}

// Transpose + convert weights to bf16 [n][k] so each lane's B fragment is two
// contiguous 16B loads.  Wqkv: f32 [256][768] -> bf16 [768][256];
// Wout: f32 [256][256] -> bf16 [256][256] (transposed).
__global__ void prep_weights(const float* __restrict__ wqkv,
                             const float* __restrict__ wout,
                             unsigned short* __restrict__ wqkvT,
                             unsigned short* __restrict__ woutT) {
    int idx = blockIdx.x * blockDim.x + threadIdx.x;
    __bf16* qT = reinterpret_cast<__bf16*>(wqkvT);
    __bf16* oT = reinterpret_cast<__bf16*>(woutT);
    if (idx < NQKV * CC) {
        int n = idx >> 8, k = idx & 255;
        qT[idx] = (__bf16)wqkv[k * NQKV + n];
    }
    if (idx < CC * CC) {
        int n = idx >> 8, k = idx & 255;
        oT[idx] = (__bf16)wout[k * CC + n];
    }
}

__global__ __launch_bounds__(256) void cav_attn_fused(
    const float* __restrict__ x, const float* __restrict__ mask,
    const unsigned short* __restrict__ wqkvT_,
    const unsigned short* __restrict__ woutT_,
    const float* __restrict__ bout, float* __restrict__ out) {
    const __bf16* wqkvT = reinterpret_cast<const __bf16*>(wqkvT_);
    const __bf16* woutT = reinterpret_cast<const __bf16*>(woutT_);

    extern __shared__ char smem[];
    __bf16* sA  = reinterpret_cast<__bf16*>(smem);            // 80 x LDA bf16
    float*  sQf = reinterpret_cast<float*>(smem + SMEM_A_BYTES); // 80 x LDQF f32 qkv
    float*  sOut = sQf;                                       // reused: 80 x LDO f32

    const int tid   = threadIdx.x;
    const int wave  = tid >> 5;
    const int lane  = tid & 31;
    const int lmod  = lane & 15;   // m (A) / n (B) / col (C,D)
    const int lhalf = lane >> 4;   // K-half select per ISA layout
    const int siteBase = blockIdx.x * SITES_PER_BLK;

    // ---- Phase 0: gather x rows (f32, scattered by (b,l,h,w)) -> bf16 LDS ----
    for (int idx = tid; idx < ROWS * 64; idx += 256) {
        int row  = idx >> 6, c4 = idx & 63;
        int site = siteBase + row / LL, l = row % LL;
        int b    = site / HWHW, hw = site % HWHW;
        const float4* src =
            (const float4*)(x + (size_t)((b * LL + l) * HWHW + hw) * CC);
        float4 v = src[c4];
        __bf16* dst = sA + row * LDA + c4 * 4;
        dst[0] = (__bf16)v.x; dst[1] = (__bf16)v.y;
        dst[2] = (__bf16)v.z; dst[3] = (__bf16)v.w;
    }
    __syncthreads();

    // ---- Phase 1: qkv(80x768,f32) = X(80x256) @ Wqkv ; 8 waves x 6 N-tiles ----
    for (int nti = 0; nti < 6; ++nti) {
        int nt = wave * 6 + nti;
        v8f acc[5] = {};
        const __bf16* bb = wqkvT + (size_t)(nt * 16 + lmod) * CC + lhalf * 8;
        for (int ks = 0; ks < 8; ++ks) {
            v16bf bf = load_frag16(bb + ks * 32);
#pragma unroll
            for (int mt = 0; mt < 5; ++mt) {
                v16bf af = load_frag16(sA + (mt * 16 + lmod) * LDA + ks * 32 + lhalf * 8);
                acc[mt] = __builtin_amdgcn_wmma_f32_16x16x32_bf16(
                    false, af, false, bf, (short)0, acc[mt], false, false);
            }
        }
#pragma unroll
        for (int mt = 0; mt < 5; ++mt)
#pragma unroll
            for (int r = 0; r < 8; ++r) {
                int row = mt * 16 + lhalf * 8 + r; // ISA C/D layout
                sQf[row * LDQF + nt * 16 + lmod] = acc[mt][r];
            }
    }
    __syncthreads();

    // ---- Phase 2: per-(site,head,i) attention (L=5, f32 VALU) -> bf16 sA ----
    for (int t = tid; t < SITES_PER_BLK * 8 * LL; t += 256) {
        int s = t / 40, rem = t % 40, head = rem / 5, i = rem % 5;
        int site = siteBase + s;
        int b = site / HWHW, hw = site % HWHW;
        int baseRow = s * LL;
        const float* qrow = sQf + (baseRow + i) * LDQF + head * 32;
        float q[32];
#pragma unroll
        for (int c = 0; c < 32; ++c) q[c] = qrow[c];
        float sc[5];
#pragma unroll
        for (int j = 0; j < 5; ++j) {
            const float* krow = sQf + (baseRow + j) * LDQF + 256 + head * 32;
            float d = 0.f;
#pragma unroll
            for (int c = 0; c < 32; ++c) d += q[c] * krow[c];
            float mv = mask[(b * HWHW + hw) * LL + j];
            sc[j] = (mv == 0.f) ? -1e30f : d * SCALEF;
        }
        float mx = sc[0];
#pragma unroll
        for (int j = 1; j < 5; ++j) mx = fmaxf(mx, sc[j]);
        float sum = 0.f;
#pragma unroll
        for (int j = 0; j < 5; ++j) { sc[j] = __expf(sc[j] - mx); sum += sc[j]; }
        float inv = 1.f / sum;
        float o[32] = {};
#pragma unroll
        for (int j = 0; j < 5; ++j) {
            const float* vrow = sQf + (baseRow + j) * LDQF + 512 + head * 32;
            float p = sc[j] * inv;
#pragma unroll
            for (int c = 0; c < 32; ++c) o[c] += p * vrow[c];
        }
        __bf16* orow = sA + (baseRow + i) * LDA + head * 32;
#pragma unroll
        for (int c = 0; c < 32; ++c) orow[c] = (__bf16)o[c];
    }
    __syncthreads();

    // ---- Phase 3: out(80x256) = attnOut @ Wout + b -> staged f32 in LDS ----
    for (int nti = 0; nti < 2; ++nti) {
        int nt = wave * 2 + nti;
        v8f acc[5] = {};
        const __bf16* bb = woutT + (size_t)(nt * 16 + lmod) * CC + lhalf * 8;
        for (int ks = 0; ks < 8; ++ks) {
            v16bf bf = load_frag16(bb + ks * 32);
#pragma unroll
            for (int mt = 0; mt < 5; ++mt) {
                v16bf af = load_frag16(sA + (mt * 16 + lmod) * LDA + ks * 32 + lhalf * 8);
                acc[mt] = __builtin_amdgcn_wmma_f32_16x16x32_bf16(
                    false, af, false, bf, (short)0, acc[mt], false, false);
            }
        }
        int col = nt * 16 + lmod;
        float bias = bout[col];
#pragma unroll
        for (int mt = 0; mt < 5; ++mt)
#pragma unroll
            for (int r = 0; r < 8; ++r) {
                int row = mt * 16 + lhalf * 8 + r;
                sOut[row * LDO + col] = acc[mt][r] + bias;
            }
    }
    __syncthreads();

    // ---- Phase 4: coalesced float4 write-out of the staged tile ----
    for (int idx = tid; idx < ROWS * 64; idx += 256) {
        int row  = idx >> 6, c4 = idx & 63;
        int site = siteBase + row / LL, l = row % LL;
        int b    = site / HWHW, hw = site % HWHW;
        float4 v = ((const float4*)(sOut + row * LDO))[c4];
        ((float4*)(out + (size_t)((b * LL + l) * HWHW + hw) * CC))[c4] = v;
    }
}

extern "C" void kernel_launch(void* const* d_in, const int* in_sizes, int n_in,
                              void* d_out, int out_size, void* d_ws, size_t ws_size,
                              hipStream_t stream) {
    const float* x    = (const float*)d_in[0];
    const float* mask = (const float*)d_in[1];
    const float* wqkv = (const float*)d_in[2];
    const float* wout = (const float*)d_in[3];
    const float* bout = (const float*)d_in[4];
    float* out = (float*)d_out;

    unsigned short* wqkvT = (unsigned short*)d_ws;      // 768*256 bf16
    unsigned short* woutT = wqkvT + (size_t)NQKV * CC;  // 256*256 bf16

    prep_weights<<<(NQKV * CC) / 256, 256, 0, stream>>>(wqkv, wout, wqkvT, woutT);
    cav_attn_fused<<<SITES / SITES_PER_BLK, 256, SMEM_BYTES, stream>>>(
        x, mask, wqkvT, woutT, bout, out);
}
